// GRU_68624987456124
// MI455X (gfx1250) — matrix operand order repaired and checked by reading, not compiled, not run on hardware
//
#include <hip/hip_runtime.h>

#define EMB  128
#define HID  256
#define G3   768          // 3*HID
#define TS   512          // timesteps
#define BATCH 256
#define NGT  48           // G3/16 gate tiles
#define HSTR 260          // padded LDS stride for h (floats)
#define GSTR 770          // padded LDS stride for hg (floats)

typedef __attribute__((ext_vector_type(16))) __bf16        v16bf;
typedef __attribute__((ext_vector_type(8)))  float         v8f;
typedef __attribute__((ext_vector_type(8)))  unsigned int  v8u;
typedef __attribute__((ext_vector_type(4)))  unsigned int  v4u;

__device__ __forceinline__ unsigned f2bf(float x) {
    unsigned u = __float_as_uint(x);
    return (u + 0x7FFFu + ((u >> 16) & 1u)) >> 16;   // RNE
}
__device__ __forceinline__ unsigned pack2(float lo, float hi) {
    return f2bf(lo) | (f2bf(hi) << 16);
}

// Load a pre-swizzled bf16 B-fragment: [32 lanes][8 vgprs] u32, 32B per lane.
__device__ __forceinline__ v16bf load_bfrag(const unsigned* __restrict__ frag, int lane) {
    v8u v = *((const v8u*)frag + lane);
    return __builtin_bit_cast(v16bf, v);
}

// Build an A-fragment (16x32 bf16) from an f32 row pointer.
// VGPR i holds K = (i>>2)*16 + kh*8 + (i&3)*2 (+1 in high 16 bits).
template <typename PTR>
__device__ __forceinline__ v16bf make_afrag(PTR rowp, int kh) {
    v8u r;
#pragma unroll
    for (int i = 0; i < 8; ++i) {
        int k = ((i >> 2) * 16) + kh * 8 + ((i & 3) * 2);
        float2 f = *(const float2*)(rowp + k);
        r[i] = pack2(f.x, f.y);
    }
    return __builtin_bit_cast(v16bf, r);
}

// ---------------------------------------------------------------------------
// Tensor Data Mover: async 2D tile load (f32 elements) Global -> LDS.
// D# packing per CDNA5 ISA. 2D: groups 2/3 zero.
// ---------------------------------------------------------------------------
__device__ __forceinline__ void tdm_load_2d(unsigned lds_off, unsigned long long gaddr,
                                            unsigned tile_d0, unsigned tile_d1,
                                            unsigned tensor_d0, unsigned tensor_d1,
                                            unsigned long long stride0_elems) {
    v4u g0 = { 1u,                                   // count=1, user mode
               lds_off,                              // lds_addr (bytes)
               (unsigned)gaddr,                      // global_addr[31:0]
               (unsigned)((gaddr >> 32) & 0x01FFFFFFull) | 0x80000000u }; // [56:32] | type=2
    v8u g1 = { 0x00020000u,                          // wg_mask=0, data_size=2 (4B)
               (tensor_d0 & 0xFFFFu) << 16,          // tensor_dim0[15:0]
               (tensor_d0 >> 16) | ((tensor_d1 & 0xFFFFu) << 16),
               (tensor_d1 >> 16) | (tile_d0 << 16),  // tile_dim0
               tile_d1,                              // tile_dim1 (tile_dim2=0)
               (unsigned)stride0_elems,              // tensor_dim0_stride lo
               (unsigned)(stride0_elems >> 32),      // stride hi; dim1_stride=0
               0u };
    v4u gz = { 0u, 0u, 0u, 0u };                     // groups 2/3: all dims unused
    asm volatile("tensor_load_to_lds %0, %1, %2, %3"
                 :: "s"(g0), "s"(g1), "s"(gz), "s"(gz)
                 : "memory");
}

// ---------------------------------------------------------------------------
// Pre-swizzle W[G3 x K] (f32) into bf16 WMMA B-fragments [frag][lane][vgpr].
// B[k][n] = W[gtile*16+n][k]  (B = W^T tile).
// ---------------------------------------------------------------------------
__global__ void swizzle_w(const float* __restrict__ W, unsigned* __restrict__ dst,
                          int K, int nkt, int total) {
    int idx = blockIdx.x * blockDim.x + threadIdx.x;
    if (idx >= total) return;
    int vg   = idx & 7;
    int lane = (idx >> 3) & 31;
    int frag = idx >> 8;
    int kt   = frag % nkt;
    int gt   = frag / nkt;
    int n  = lane & 15;
    int kh = lane >> 4;
    int k  = kt * 32 + (vg >> 2) * 16 + kh * 8 + (vg & 3) * 2;
    const float* row = W + (size_t)(gt * 16 + n) * K;
    dst[idx] = pack2(row[k], row[k + 1]);
}

// ---------------------------------------------------------------------------
// xg = x @ W_ih^T + b_ih  (M=131072, K=128, N=768). Each wave: two 16-row
// M tiles (B-fragment reuse x2), all 48 gate tiles, bias folded into stores.
// ---------------------------------------------------------------------------
__global__ __launch_bounds__(256) void gemm_xg(
    const float* __restrict__ x, const float* __restrict__ b_ih,
    const unsigned* __restrict__ swWih, float* __restrict__ xg) {
    int wave = threadIdx.x >> 5;
    int lane = threadIdx.x & 31;
    int mt0  = (blockIdx.x * 8 + wave) * 2;          // two consecutive M tiles
    int kh   = lane >> 4;

    v16bf a[2][4];
#pragma unroll
    for (int mi = 0; mi < 2; ++mi) {
        const float* rowp = x + (size_t)(((mt0 + mi) << 4) + (lane & 15)) * EMB;
#pragma unroll
        for (int kt = 0; kt < 4; ++kt) a[mi][kt] = make_afrag(rowp + kt * 32, kh);
    }

#pragma unroll 1
    for (int gt = 0; gt < NGT; ++gt) {
        float bias = b_ih[gt * 16 + (lane & 15)];
        v8f acc0 = {}, acc1 = {};
#pragma unroll
        for (int kt = 0; kt < 4; ++kt) {
            v16bf b = load_bfrag(swWih + ((gt * 4 + kt) << 8), lane);
            acc0 = __builtin_amdgcn_wmma_f32_16x16x32_bf16(
                false, a[0][kt], false, b, (short)0, acc0, false, false);
            acc1 = __builtin_amdgcn_wmma_f32_16x16x32_bf16(
                false, a[1][kt], false, b, (short)0, acc1, false, false);
        }
        size_t col = gt * 16 + (lane & 15);
        size_t b0 = (size_t)(mt0 * 16 + kh * 8) * G3 + col;
        size_t b1 = (size_t)((mt0 + 1) * 16 + kh * 8) * G3 + col;
#pragma unroll
        for (int r = 0; r < 8; ++r) {
            xg[b0 + (size_t)r * G3] = acc0[r] + bias;
            xg[b1 + (size_t)r * G3] = acc1[r] + bias;
        }
    }
}

// ---------------------------------------------------------------------------
// Persistent GRU scan: 16 WGs, one 16-row batch tile each, 512 steps.
// Per step: hg = h @ W_hh^T + b_hh (bf16 WMMA, 8 waves x 6 gate tiles);
// xg gate inputs double-buffered in LDS via TDM tensor_load_to_lds,
// overlapped with compute and synced by s_wait_tensorcnt.
// ---------------------------------------------------------------------------
__global__ __launch_bounds__(256) void gru_scan(
    const float* __restrict__ h0, const float* __restrict__ b_hh,
    const unsigned* __restrict__ swWhh, const float* __restrict__ xg,
    const float* __restrict__ fc_w, const float* __restrict__ fc_b,
    float* __restrict__ out) {
    extern __shared__ float smem[];
    float* h    = smem;                          // [16][HSTR]
    float* hg   = smem + 16 * HSTR;              // [16][GSTR]
    float* xbuf = smem + 16 * HSTR + 16 * GSTR;  // [2][16][G3] TDM staging

    int tid  = threadIdx.x;
    int wave = tid >> 5;
    int lane = tid & 31;
    int bt   = blockIdx.x;                       // batch tile 0..15
    int mrow = lane & 15;
    int kh   = lane >> 4;

    for (int i = tid; i < 16 * HID; i += 256) {
        int r = i >> 8, c = i & 255;
        h[r * HSTR + c] = h0[(size_t)(bt * 16 + r) * HID + c];
    }
    float bias[6];
#pragma unroll
    for (int g = 0; g < 6; ++g) bias[g] = b_hh[(wave * 6 + g) * 16 + (lane & 15)];

    // kick off TDM for step 0 (wave 0 owns the TENSORcnt stream)
    if (wave == 0) {
        unsigned lds_off = (unsigned)(size_t)(void*)&xbuf[0];
        unsigned long long ga = (unsigned long long)(size_t)xg
                              + (unsigned long long)(bt * 16) * TS * G3 * 4ull;
        tdm_load_2d(lds_off, ga, G3, 16, G3, 16, (unsigned long long)TS * G3);
    }
    __syncthreads();

    for (int t = 0; t < TS; ++t) {
        // A fragments of h (f32 LDS -> bf16), shared across this wave's tiles
        v16bf a[8];
#pragma unroll
        for (int kt = 0; kt < 8; ++kt)
            a[kt] = make_afrag(&h[mrow * HSTR + kt * 32], kh);

#pragma unroll 1
        for (int g = 0; g < 6; ++g) {
            int gt = wave * 6 + g;
            v8f acc;
#pragma unroll
            for (int i = 0; i < 8; ++i) acc[i] = bias[g];
#pragma unroll
            for (int kt = 0; kt < 8; ++kt) {
                v16bf b = load_bfrag(swWhh + ((gt * 8 + kt) << 8), lane);
                acc = __builtin_amdgcn_wmma_f32_16x16x32_bf16(
                    false, a[kt], false, b, (short)0, acc, false, false);
            }
            int col = gt * 16 + (lane & 15);
#pragma unroll
            for (int r = 0; r < 8; ++r) hg[(kh * 8 + r) * GSTR + col] = acc[r];
        }
        __builtin_amdgcn_s_wait_tensorcnt(0);    // xbuf[t&1] resident
        __syncthreads();                         // hg + TDM data visible to all

        // launch TDM for step t+1 into the other buffer (consumed two barriers later)
        if (wave == 0 && t + 1 < TS) {
            unsigned lds_off = (unsigned)(size_t)(void*)&xbuf[((t + 1) & 1) * (16 * G3)];
            unsigned long long ga = (unsigned long long)(size_t)xg
                + ((unsigned long long)(bt * 16) * TS + (t + 1)) * G3 * 4ull;
            tdm_load_2d(lds_off, ga, G3, 16, G3, 16, (unsigned long long)TS * G3);
        }

        // gate epilogue: thread tid owns hidden column j = tid for all 16 rows
        const float* xb = xbuf + (t & 1) * (16 * G3);
        int j = tid;
#pragma unroll 1
        for (int r = 0; r < 16; ++r) {
            float xr = xb[r * G3 + j];
            float xz = xb[r * G3 + HID + j];
            float xn = xb[r * G3 + 2 * HID + j];
            float hr = hg[r * GSTR + j];
            float hz = hg[r * GSTR + HID + j];
            float hn = hg[r * GSTR + 2 * HID + j];
            float rg = 1.f / (1.f + __expf(-(xr + hr)));
            float zg = 1.f / (1.f + __expf(-(xz + hz)));
            float ng = tanhf(xn + rg * hn);
            float hp = h[r * HSTR + j];
            h[r * HSTR + j] = (1.f - zg) * ng + zg * hp;
        }
        __syncthreads();
    }

    // outputs: y_hat[0..255], then hT[256 x 256]
    for (int r = 0; r < 16; ++r)
        out[BATCH + (size_t)(bt * 16 + r) * HID + tid] = h[r * HSTR + tid];
    if (tid < 16) {
        float s = 0.f;
        for (int k = 0; k < HID; ++k) s += h[tid * HSTR + k] * fc_w[k];
        out[bt * 16 + tid] = s + fc_b[0];
    }
}

extern "C" void kernel_launch(void* const* d_in, const int* in_sizes, int n_in,
                              void* d_out, int out_size, void* d_ws, size_t ws_size,
                              hipStream_t stream) {
    const float* x   = (const float*)d_in[0];
    const float* h0  = (const float*)d_in[1];
    const float* Wih = (const float*)d_in[2];
    const float* Whh = (const float*)d_in[3];
    const float* bih = (const float*)d_in[4];
    const float* bhh = (const float*)d_in[5];
    const float* fcw = (const float*)d_in[6];
    const float* fcb = (const float*)d_in[7];
    float* out = (float*)d_out;

    const size_t M = (size_t)BATCH * TS;                 // 131072 rows
    float*    xg    = (float*)d_ws;                      // M x 768 f32
    unsigned* swWih = (unsigned*)((char*)d_ws + M * G3 * sizeof(float));
    unsigned* swWhh = swWih + NGT * 4 * 256;             // after 192KB of W_ih frags

    int totIh = NGT * 4 * 256;   // 49152 u32
    int totHh = NGT * 8 * 256;   // 98304 u32
    swizzle_w<<<(totIh + 255) / 256, 256, 0, stream>>>(Wih, swWih, EMB, 4, totIh);
    swizzle_w<<<(totHh + 255) / 256, 256, 0, stream>>>(Whh, swWhh, HID, 8, totHh);

    gemm_xg<<<(int)(M / 16 / 16), 256, 0, stream>>>(x, bih, swWih, xg);

    size_t shbytes = (size_t)(16 * HSTR + 16 * GSTR + 2 * 16 * G3) * sizeof(float);
    gru_scan<<<16, 256, shbytes, stream>>>(h0, bhh, swWhh, xg, fcw, fcb, out);
}
